// TensorProductConvLayer_45732811768272
// MI455X (gfx1250) — matrix operand order repaired and checked by reading, not compiled, not run on hardware
//
#include <hip/hip_runtime.h>
#include <math.h>

// ---------------- problem constants ----------------
#define N_NODES   16384
#define N_EDGES   131072
#define MUL       16
#define NEF       64
#define W_NUMEL   1024          // 4*MUL*MUL
#define INV_SQRT3 0.57735026918962576f
#define ALPHA     0.17677669529663687f   // 1/sqrt(2*MUL)

// workspace layout (bytes)
#define PW1_OFF   0u                       // 4096  bf16  (8 KB)
#define PW2_OFF   8192u                    // 65536 bf16  (128 KB)
#define SUMS_OFF  139264u                  // 1048576 f32 (4 MB)
#define CNT_OFF   4333568u                 // 16384 f32   (64 KB)

typedef __attribute__((ext_vector_type(16))) __bf16 v16bf;
typedef __attribute__((ext_vector_type(8)))  __bf16 v8bf;
typedef __attribute__((ext_vector_type(8)))  float  v8f;

__device__ __forceinline__ unsigned short f2bf(float f) {
    unsigned u = __float_as_uint(f);
    u += 0x7fffu + ((u >> 16) & 1u);          // round to nearest even
    return (unsigned short)(u >> 16);
}

// branchless, bf16-accurate softplus: max(x,0) + log(1 + exp(-|x|))
__device__ __forceinline__ float softplus_fast(float x) {
    return fmaxf(x, 0.0f) + __logf(1.0f + __expf(-fabsf(x)));
}

// ------------------------------------------------------------------
// Kernel 0: zero scatter buffers + pack fc1_w / fc2_w into the exact
// per-lane WMMA B-fragment layout (bf16).
//   element i of pack = (((tile*2 + kstep)*32 + lane)*16 + ii)
//   n = tile*16 + (lane&15),  k = 32*kstep + 8*(lane>>4) + (ii&7) + 16*(ii>>3)
// ------------------------------------------------------------------
__global__ __launch_bounds__(256) void k_init_pack(
    const float* __restrict__ fc1_w, const float* __restrict__ fc2_w,
    unsigned short* __restrict__ pw1, unsigned short* __restrict__ pw2,
    float* __restrict__ sums, float* __restrict__ cnt)
{
    const unsigned i = blockIdx.x * 256u + threadIdx.x;   // 0 .. 1048575
    sums[i] = 0.0f;
    if (i < N_NODES) cnt[i] = 0.0f;
    if (i < 65536u) {
        unsigned ii = i & 15u;
        unsigned L  = (i >> 4) & 31u;
        unsigned s  = (i >> 9) & 1u;
        unsigned t  =  i >> 10;
        unsigned n  = t * 16u + (L & 15u);
        unsigned k  = 32u * s + 8u * (L >> 4) + (ii & 7u) + 16u * (ii >> 3);
        pw2[i] = f2bf(fc2_w[k * W_NUMEL + n]);
    }
    if (i < 4096u) {
        unsigned ii = i & 15u;
        unsigned L  = (i >> 4) & 31u;
        unsigned s  = (i >> 9) & 1u;
        unsigned t  =  i >> 10;            // 0..3
        unsigned n  = t * 16u + (L & 15u);
        unsigned k  = 32u * s + 8u * (L >> 4) + (ii & 7u) + 16u * (ii >> 3);
        pw1[i] = f2bf(fc1_w[k * NEF + n]);
    }
}

// ------------------------------------------------------------------
// Kernel 1: fused MLP (WMMA bf16) + tensor product + atomic scatter.
// One block = 16 edges, 256 threads = 8 wave32.
// ------------------------------------------------------------------
__global__ __launch_bounds__(256) void k_edge_fused(
    const float* __restrict__ node_attr,
    const long long* __restrict__ edge_index,
    const float* __restrict__ edge_attr,
    const float* __restrict__ edge_sh,
    const float* __restrict__ fc1_b,
    const float* __restrict__ fc2_b,
    const unsigned short* __restrict__ pw1r,
    const unsigned short* __restrict__ pw2r,
    float* __restrict__ sums, float* __restrict__ cnt)
{
    __shared__ unsigned short ebuf[16 * 64];   // edge_attr tile, bf16, row-major
    __shared__ unsigned short hbuf[16 * 64];   // softplus hidden, bf16, row-major
    __shared__ float wbuf[16 * 1024];          // GEMM2 output (f32), 64 KB
    __shared__ float na[16 * 64];              // gathered node_attr[dst]
    __shared__ float shv[16 * 4];              // edge_sh tile
    __shared__ float dotb[16 * 16];            // dot_u = (vd . y1) * INV_SQRT3
    __shared__ int   srcL[16], dstL[16];

    const int tid  = threadIdx.x;
    const int lane = tid & 31;
    const int wave = tid >> 5;
    const int kh   = lane >> 4;      // K-half select (A/B fragment layout)
    const int nrow = lane & 15;      // M for A-frag / N for B,C,D frags
    const int e0   = blockIdx.x * 16;

    const __bf16* pw1 = (const __bf16*)pw1r;
    const __bf16* pw2 = (const __bf16*)pw2r;

    // warm the B-pack in cache (global_prefetch)
    __builtin_prefetch(&pw2r[(size_t)tid * 256u], 0, 1);

    if (tid < 16) {
        srcL[tid] = (int)edge_index[e0 + tid];
        dstL[tid] = (int)edge_index[N_EDGES + e0 + tid];
    }
    if (tid < 64) shv[tid] = edge_sh[(size_t)e0 * 4 + tid];
    __syncthreads();

    // edge_attr tile -> bf16 LDS (each thread converts 4 floats)
    {
        const float4 f = ((const float4*)(edge_attr + (size_t)e0 * NEF))[tid];
        ebuf[tid * 4 + 0] = f2bf(f.x);
        ebuf[tid * 4 + 1] = f2bf(f.y);
        ebuf[tid * 4 + 2] = f2bf(f.z);
        ebuf[tid * 4 + 3] = f2bf(f.w);
    }
    // gather node_attr[dst] rows
    for (int i = tid; i < 16 * 64; i += 256) {
        int e = i >> 6, c = i & 63;
        na[i] = node_attr[(size_t)dstL[e] * 64 + c];
    }
    __syncthreads();

    // per-(edge,u) vector dot with y1
    {
        int e = tid >> 4, u = tid & 15;
        const float* vv = &na[e * 64 + 16 + u * 3];
        dotb[tid] = INV_SQRT3 *
            (vv[0] * shv[e * 4 + 1] + vv[1] * shv[e * 4 + 2] + vv[2] * shv[e * 4 + 3]);
    }

    // ---------------- GEMM1: h = softplus(E @ W1 + b1)  (waves 0..3) -------
    if (wave < 4) {
        v8f acc = {};
#pragma unroll
        for (int s = 0; s < 2; ++s) {
            union { v16bf v; v8bf h[2]; } a;
            a.h[0] = *(const v8bf*)&ebuf[nrow * 64 + s * 32 + kh * 8];
            a.h[1] = *(const v8bf*)&ebuf[nrow * 64 + s * 32 + kh * 8 + 16];
            v16bf b = *(const v16bf*)&pw1[(size_t)(((wave * 2 + s) * 32 + lane)) * 16];
            acc = __builtin_amdgcn_wmma_f32_16x16x32_bf16(
                      false, a.v, false, b, (short)0, acc, false, false);
        }
        const float bb = fc1_b[wave * 16 + nrow];
#pragma unroll
        for (int r = 0; r < 8; ++r) {
            float h = softplus_fast(acc[r] + bb);
            hbuf[(r + 8 * kh) * 64 + wave * 16 + nrow] = f2bf(h);
        }
    }
    __syncthreads();

    // ---------------- GEMM2: w = h @ W2 + b2  (all 8 waves, 8 N-tiles each)
    v8f acc[8];
#pragma unroll
    for (int t = 0; t < 8; ++t) acc[t] = (v8f){};
#pragma unroll
    for (int s = 0; s < 2; ++s) {
        union { v16bf v; v8bf h[2]; } a;
        a.h[0] = *(const v8bf*)&hbuf[nrow * 64 + s * 32 + kh * 8];
        a.h[1] = *(const v8bf*)&hbuf[nrow * 64 + s * 32 + kh * 8 + 16];
#pragma unroll
        for (int t = 0; t < 8; ++t) {
            const int tg = wave * 8 + t;
            v16bf b = *(const v16bf*)&pw2[(size_t)((tg * 2 + s) * 32 + lane) * 16];
            acc[t] = __builtin_amdgcn_wmma_f32_16x16x32_bf16(
                         false, a.v, false, b, (short)0, acc[t], false, false);
        }
    }
#pragma unroll
    for (int t = 0; t < 8; ++t) {
        const int col = (wave * 8 + t) * 16 + nrow;
        const float bb = fc2_b[col];
#pragma unroll
        for (int r = 0; r < 8; ++r)
            wbuf[(r + 8 * kh) * 1024 + col] = acc[t][r] + bb;
    }
    __syncthreads();

    // ---------------- tensor product + atomic scatter ----------------------
    for (int i = tid; i < 16 * 64; i += 256) {
        const int e = i >> 6, j = i & 63;
        const float* w = &wbuf[e * 1024];
        const float y0 = shv[e * 4 + 0];
        float val;
        if (j < 16) {
            const int wo = j;
            float acs = 0.0f;
#pragma unroll
            for (int u = 0; u < 16; ++u)
                acs += (na[e * 64 + u] * y0) * w[u * 16 + wo]
                     +  dotb[e * 16 + u]    * w[256 + u * 16 + wo];
            val = ALPHA * acs;
        } else {
            const int jj = j - 16;
            const int wo = jj / 3, k = jj - wo * 3;
            const float y1k = shv[e * 4 + 1 + k];
            float acc_c = 0.0f, acc_d = 0.0f;
#pragma unroll
            for (int u = 0; u < 16; ++u) {
                acc_c += na[e * 64 + u]              * w[512 + u * 16 + wo];
                acc_d += na[e * 64 + 16 + u * 3 + k] * w[768 + u * 16 + wo];
            }
            val = ALPHA * (acc_c * y1k + acc_d * y0);
        }
        atomicAdd(&sums[(size_t)srcL[e] * 64 + j], val);
    }
    if (tid < 16) atomicAdd(&cnt[srcL[tid]], 1.0f);
}

// ------------------------------------------------------------------
// Kernel 2: out = sums / max(cnt,1) + node_attr
// ------------------------------------------------------------------
__global__ __launch_bounds__(256) void k_finalize(
    const float* __restrict__ sums, const float* __restrict__ cnt,
    const float* __restrict__ node_attr, float* __restrict__ out)
{
    const unsigned i = blockIdx.x * 256u + threadIdx.x;   // 0 .. 1048575
    out[i] = sums[i] / fmaxf(cnt[i >> 6], 1.0f) + node_attr[i];
}

// ------------------------------------------------------------------
extern "C" void kernel_launch(void* const* d_in, const int* in_sizes, int n_in,
                              void* d_out, int out_size, void* d_ws, size_t ws_size,
                              hipStream_t stream)
{
    const float*     node_attr  = (const float*)d_in[0];
    const long long* edge_index = (const long long*)d_in[1];   // int64 per reference
    const float*     edge_attr  = (const float*)d_in[2];
    const float*     edge_sh    = (const float*)d_in[3];
    const float*     fc1_w      = (const float*)d_in[4];
    const float*     fc1_b      = (const float*)d_in[5];
    const float*     fc2_w      = (const float*)d_in[6];
    const float*     fc2_b      = (const float*)d_in[7];

    char* ws = (char*)d_ws;
    unsigned short* pw1  = (unsigned short*)(ws + PW1_OFF);
    unsigned short* pw2  = (unsigned short*)(ws + PW2_OFF);
    float*          sums = (float*)(ws + SUMS_OFF);
    float*          cnt  = (float*)(ws + CNT_OFF);
    float*          out  = (float*)d_out;

    // 1M threads: zero sums/cnt, pack weights to bf16 B-fragment layout
    k_init_pack<<<(N_NODES * 64) / 256, 256, 0, stream>>>(fc1_w, fc2_w, pw1, pw2, sums, cnt);

    // 16 edges per block
    k_edge_fused<<<N_EDGES / 16, 256, 0, stream>>>(
        node_attr, edge_index, edge_attr, edge_sh, fc1_b, fc2_b, pw1, pw2, sums, cnt);

    // segment mean + residual
    k_finalize<<<(N_NODES * 64) / 256, 256, 0, stream>>>(sums, cnt, node_attr, out);
}
